// SprecherLayerBlock_39298950758545
// MI455X (gfx1250) — matrix operand order repaired and checked by reading, not compiled, not run on hardware
//
#include <hip/hip_runtime.h>
#include <hip/hip_bf16.h>
#include <math.h>

typedef __attribute__((ext_vector_type(16))) _Float16 v16h;
typedef __attribute__((ext_vector_type(8)))  float    v8f;

#define BB 1024
#define PP 256
#define QQ 256
#define KK 256

// ---------------------------------------------------------------------------
// Phase A: build spline coefficient tables in workspace.
//   ws[0..511]   as float2[256]: phi pair table (c[i], c[i+1])
//   ws[512..767] : Phi_c (outer spline coeffs, cc/cr rescaled)
// Single block, 256 threads.
// ---------------------------------------------------------------------------
__global__ __launch_bounds__(256)
void sprecher_prep_kernel(const float* __restrict__ phi_log_inc,
                          const float* __restrict__ Phi_coeffs,
                          const float* __restrict__ cc_p,
                          const float* __restrict__ cr_p,
                          float* __restrict__ tables)
{
    __shared__ float buf[KK];
    __shared__ float phic_sh[KK];
    const int tid = threadIdx.x;

    // softplus (numerically stable, matches jax.nn.softplus)
    float z  = phi_log_inc[tid];
    float sp = fmaxf(z, 0.0f) + log1pf(expf(-fabsf(z)));

    // inclusive cumsum via Hillis-Steele scan in LDS
    buf[tid] = sp;
    __syncthreads();
    #pragma unroll
    for (int off = 1; off < KK; off <<= 1) {
        float v = (tid >= off) ? buf[tid - off] : 0.0f;
        __syncthreads();
        buf[tid] += v;
        __syncthreads();
    }
    float cum   = buf[tid];
    float total = buf[KK - 1];
    phic_sh[tid] = cum / (total + 1e-8f);
    __syncthreads();

    // emit adjacent-pair table (one ds_load_b64 per gather in main kernel)
    float2 pr;
    pr.x = phic_sh[tid];
    pr.y = phic_sh[(tid == KK - 1) ? tid : tid + 1];
    ((float2*)tables)[tid] = pr;

    // min/max reduction of Phi_coeffs
    float pv = Phi_coeffs[tid];
    buf[tid] = pv;
    __syncthreads();
    #pragma unroll
    for (int off = 128; off > 0; off >>= 1) {
        if (tid < off) buf[tid] = fminf(buf[tid], buf[tid + off]);
        __syncthreads();
    }
    float cmin = buf[0];
    __syncthreads();
    buf[tid] = pv;
    __syncthreads();
    #pragma unroll
    for (int off = 128; off > 0; off >>= 1) {
        if (tid < off) buf[tid] = fmaxf(buf[tid], buf[tid + off]);
        __syncthreads();
    }
    float cmax = buf[0];

    float cc = cc_p[0];
    float cr = cr_p[0];
    float norm = (pv - cmin) / (cmax - cmin + 1e-8f);
    tables[2 * KK + tid] = cc - cr + 2.0f * cr * norm;
}

// ---------------------------------------------------------------------------
// Inner spline eval entirely in u-space (knot-index units).
// Input u0 = x*invh + (shift - k0)*invh, precombined outside.
//   u = clamp(u0, 0, K-1) ; fi = min(floor(u), K-2) ; t = u - fi
// Pair-table gather: single ds_load_b64.
// ---------------------------------------------------------------------------
__device__ __forceinline__ float phi_eval_u(float u0, const float2* tab)
{
    float u  = fminf(fmaxf(u0, 0.0f), (float)(KK - 1));
    float fi = fminf(floorf(u), (float)(KK - 2));
    float t  = u - fi;
    float2 pr = tab[(int)fi];
    return fmaf(t, pr.y - pr.x, pr.x);
}

// ---------------------------------------------------------------------------
// Phase B: main Sprecher layer. One block per batch row b (512 thr = 16 waves).
// Wave w computes s[q], q in [16w,16w+16):
//   A[m,k] (f16) = phi(x[p0+k] + shift[16w+m])      (M rows = q, K = p chunk)
//   B[k,n] (f16) = lambda[p0+k]  (broadcast over n; contiguous 16-f16 LDS read)
//   C += A*B  via V_WMMA_F32_16X16X32_F16 -> every C column equals s[q-row].
// x is staged PRE-SCALED by invh so the inner loop needs only one add per
// element to reach u-space. Then 256 threads apply the outer Phi spline.
// ---------------------------------------------------------------------------
__global__ __launch_bounds__(512)
void sprecher_main_kernel(const float* __restrict__ x,
                          const float* __restrict__ lambdas,
                          const float* __restrict__ phi_knots,
                          const float* __restrict__ Phi_knots,
                          const float* __restrict__ shifts,
                          const float* __restrict__ tables,
                          float* __restrict__ y)
{
    __shared__ float2 xs_lds[PP / 2];            // x row * invh, as pairs
    __shared__ float2 phip_lds[KK];              // phi coeff pair table
    __shared__ float  Phic_lds[KK];              // outer spline coeffs
    __shared__ __align__(32) _Float16 lam_h[PP]; // lambdas as f16 (B frags)
    __shared__ float  s_lds[QQ];

    const int b   = blockIdx.x;
    const int tid = threadIdx.x;

    // uniform inner-knot parameters (phi_knots = linspace(-4, 5, K))
    const float pk0   = phi_knots[0];
    const float pkN   = phi_knots[KK - 1];
    const float pinvh = (float)(KK - 1) / (pkN - pk0);

    if (tid < 128) {
        float2 xv = ((const float2*)(x + b * PP))[tid];
        xv.x *= pinvh;
        xv.y *= pinvh;
        xs_lds[tid] = xv;
    }
    if (tid < 256) {
        phip_lds[tid] = ((const float2*)tables)[tid];
        Phic_lds[tid] = tables[2 * KK + tid];
        lam_h[tid]    = (_Float16)lambdas[tid];
    }

    const int wave = tid >> 5;     // 0..15  (wave32)
    const int lane = tid & 31;
    const int m    = lane & 15;    // A row / C column within 16x16 tile
    const int hi   = lane >> 4;    // lane half selects K sub-range
    // per-lane shift, pre-mapped to u-space: u = x*invh + sc
    const float sc = (shifts[wave * 16 + m] - pk0) * pinvh;

    __syncthreads();

    v8f acc = {};

    #pragma unroll 4
    for (int p0 = 0; p0 < PP; p0 += 32) {
        // B fragment: 16 contiguous f16 lambdas at p0 + 16*hi (32B aligned)
        const v16h bfrag = *(const v16h*)&lam_h[p0 + (hi << 4)];

        v16h a;
        #pragma unroll
        for (int v = 0; v < 8; ++v) {
            // 16-bit A 16x32 layout: VGPR v holds K = kb, kb+1 with
            //   kb = 16*hi + 2*v   (kb is even; contiguous, matches B)
            const int kb = (hi << 4) + (v << 1);
            const float2 xv = xs_lds[(p0 + kb) >> 1];
            a[2 * v]     = (_Float16)phi_eval_u(xv.x + sc, phip_lds);
            a[2 * v + 1] = (_Float16)phi_eval_u(xv.y + sc, phip_lds);
        }
        acc = __builtin_amdgcn_wmma_f32_16x16x32_f16(
                  false, a, false, bfrag, (short)0, acc, false, false);
    }

    // Extract column N=0: lane 0 holds rows 0..7 (VGPR r -> M=r),
    // lane 16 holds rows 8..15 (VGPR r -> M=8+r).
    if (m == 0) {
        #pragma unroll
        for (int r = 0; r < 8; ++r) {
            s_lds[wave * 16 + hi * 8 + r] = acc[r];
        }
    }
    __syncthreads();

    // Outer Phi spline on uniform knots (Phi_knots = linspace(0, P, K))
    if (tid < 256) {
        const float s    = s_lds[tid];
        const float Pk0  = Phi_knots[0];
        const float PkN  = Phi_knots[KK - 1];
        const float Pinv = (float)(KK - 1) / (PkN - Pk0);

        float u  = fminf(fmaxf((s - Pk0) * Pinv, 0.0f), (float)(KK - 1));
        float fi = fminf(floorf(u), (float)(KK - 2));
        float t  = u - fi;
        int  idx = (int)fi;
        const float c0 = Phic_lds[idx];
        const float c1 = Phic_lds[idx + 1];
        y[b * QQ + tid] = fmaf(t, c1 - c0, c0);
    }
}

// ---------------------------------------------------------------------------
extern "C" void kernel_launch(void* const* d_in, const int* in_sizes, int n_in,
                              void* d_out, int out_size, void* d_ws, size_t ws_size,
                              hipStream_t stream) {
    const float* x            = (const float*)d_in[0];
    const float* lambdas      = (const float*)d_in[1];
    const float* phi_log_inc  = (const float*)d_in[2];
    const float* Phi_coeffs   = (const float*)d_in[3];
    const float* cc           = (const float*)d_in[4];
    const float* cr           = (const float*)d_in[5];
    const float* phi_knots    = (const float*)d_in[6];
    const float* Phi_knots    = (const float*)d_in[7];
    const float* shifts       = (const float*)d_in[8];
    float*       y            = (float*)d_out;
    float*       tables       = (float*)d_ws;   // 768 floats used

    sprecher_prep_kernel<<<dim3(1), dim3(256), 0, stream>>>(
        phi_log_inc, Phi_coeffs, cc, cr, tables);

    sprecher_main_kernel<<<dim3(BB), dim3(512), 0, stream>>>(
        x, lambdas, phi_knots, Phi_knots, shifts, tables, y);
}
